// VQVAE_35476429865938
// MI455X (gfx1250) — matrix operand order repaired and checked by reading, not compiled, not run on hardware
//
#include <hip/hip_runtime.h>
#include <math.h>

// ---------------------------------------------------------------------------
// VQ-VAE forward for MI455X (gfx1250, wave32).
// GEMMs: V_WMMA_F32_16X16X32_BF16 with f32 accumulation, LDS-staged
// double-buffered tiles (global loads for tile k+1 overlap WMMAs of tile k).
// ---------------------------------------------------------------------------

typedef __bf16 bf16_t;
typedef bf16_t v16bf __attribute__((ext_vector_type(16)));
typedef bf16_t v8bf  __attribute__((ext_vector_type(8)));
typedef float  v8f   __attribute__((ext_vector_type(8)));

#define BATCH   16384
#define IN_DIM  1024
#define HID     2048
#define LAT     256
#define NCODE   8192

#define LSTR 40  // LDS row stride in bf16 (32 data + 8 pad -> 80B rows, no bank conflicts)

#define WMMA_BF16(a, b, c) \
  __builtin_amdgcn_wmma_f32_16x16x32_bf16(false, (a), false, (b), (short)0, (c), false, false)

// --------------------------- LDS fragment loaders --------------------------
// LDS tiles are stored as 128 rows x 32 K-elements (row stride LSTR).
// A fragment (16x32, lane m = lane%16): half==0 lanes hold K {0..7,16..23},
// half==1 lanes hold K {8..15,24..31} (CDNA5 16-bit A layout).
__device__ inline v16bf lds_a_frag(const bf16_t* __restrict__ ldsA, int mlocal, int half) {
  union { v16bf v; v8bf h[2]; } u;
  const bf16_t* p = ldsA + mlocal * LSTR + half * 8;
  u.h[0] = *(const v8bf*)(p);
  u.h[1] = *(const v8bf*)(p + 16);
  return u.v;
}
// B fragment (32x16, lane n = lane%16): half==0 lanes hold K {0..7,16..23},
// half==1 lanes hold K {8..15,24..31}.
__device__ inline v16bf lds_b_frag(const bf16_t* __restrict__ ldsB, int nlocal, int half) {
  union { v16bf v; v8bf h[2]; } u;
  const bf16_t* p = ldsB + nlocal * LSTR + half * 8;
  u.h[0] = *(const v8bf*)(p);
  u.h[1] = *(const v8bf*)(p + 16);
  return u.v;
}

// --------------------------- tile staging ----------------------------------
// Each of the 256 threads moves one 32-byte chunk of A and one of B:
// thread t -> row r = t>>1, K-offset kh = (t&1)*16.
struct Stage { uint4 a0, a1, b0, b1; };

__device__ inline Stage stage_load(const bf16_t* __restrict__ A,
                                   const bf16_t* __restrict__ Bt, int K,
                                   int bm, int bn, int k0, int tid) {
  const int r = tid >> 1, kh = (tid & 1) * 16;
  const bf16_t* sa = A + (size_t)(bm + r) * K + k0 + kh;
  const bf16_t* sb = Bt + (size_t)(bn + r) * K + k0 + kh;
  Stage s;
  s.a0 = *(const uint4*)(sa);
  s.a1 = *(const uint4*)(sa + 8);
  s.b0 = *(const uint4*)(sb);
  s.b1 = *(const uint4*)(sb + 8);
  return s;
}

__device__ inline void stage_store(bf16_t* __restrict__ ldsA, bf16_t* __restrict__ ldsB,
                                   const Stage& s, int tid) {
  const int r = tid >> 1, kh = (tid & 1) * 16;
  *(uint4*)(ldsA + r * LSTR + kh)     = s.a0;
  *(uint4*)(ldsA + r * LSTR + kh + 8) = s.a1;
  *(uint4*)(ldsB + r * LSTR + kh)     = s.b0;
  *(uint4*)(ldsB + r * LSTR + kh + 8) = s.b1;
}

// --------------------------- WMMA micro-step -------------------------------
__device__ inline void mma_step_lds(v8f acc[2][4], const bf16_t* __restrict__ ldsA,
                                    const bf16_t* __restrict__ ldsB,
                                    int wm, int wn, int l16, int half) {
  v16bf a0 = lds_a_frag(ldsA, 32 * wm + l16, half);
  v16bf a1 = lds_a_frag(ldsA, 32 * wm + 16 + l16, half);
#pragma unroll
  for (int ni = 0; ni < 4; ++ni) {
    v16bf b = lds_b_frag(ldsB, 64 * wn + 16 * ni + l16, half);
    acc[0][ni] = WMMA_BF16(a0, b, acc[0][ni]);
    acc[1][ni] = WMMA_BF16(a1, b, acc[1][ni]);
  }
}

// --------------------------- pipelined K loop ------------------------------
// smem layout: [A0][B0][A1][B1], each 128*LSTR bf16 (40 KB total).
// Iteration kk: issue global loads for tile kk, run WMMAs on tile kk-1 from
// LDS, then ds_store tile kk (the loadcnt wait lands after the WMMA burst).
__device__ inline void gemm_k_loop(const bf16_t* __restrict__ A,
                                   const bf16_t* __restrict__ Bt, int K,
                                   int bm, int bn, bf16_t* __restrict__ smem,
                                   v8f acc[2][4], int tid) {
  const int lane = tid & 31, l16 = lane & 15, half = lane >> 4;
  const int wave = tid >> 5, wn = wave & 1, wm = wave >> 1;
  bf16_t* aBuf[2] = { smem,                  smem + 2 * 128 * LSTR };
  bf16_t* bBuf[2] = { smem + 128 * LSTR,     smem + 3 * 128 * LSTR };
  const int nK = K >> 5;

  __syncthreads();  // protect smem reuse across consecutive calls
  Stage st = stage_load(A, Bt, K, bm, bn, 0, tid);
  stage_store(aBuf[0], bBuf[0], st, tid);
  __syncthreads();

  for (int kk = 1; kk < nK; ++kk) {
    st = stage_load(A, Bt, K, bm, bn, kk << 5, tid);  // in flight during WMMAs
    if (kk + 2 < nK) {  // pull lines 2 tiles ahead into the WGP cache
      const int r = tid >> 1, kh = (tid & 1) * 16;
      __builtin_prefetch(A + (size_t)(bm + r) * K + ((kk + 2) << 5) + kh, 0, 1);
      __builtin_prefetch(Bt + (size_t)(bn + r) * K + ((kk + 2) << 5) + kh, 0, 1);
    }
    const int pb = (kk - 1) & 1, cb = kk & 1;
    mma_step_lds(acc, aBuf[pb], bBuf[pb], wm, wn, l16, half);
    stage_store(aBuf[cb], bBuf[cb], st, tid);
    __syncthreads();
  }
  mma_step_lds(acc, aBuf[(nK - 1) & 1], bBuf[(nK - 1) & 1], wm, wn, l16, half);
}

// --------------------------- generic WMMA GEMM -----------------------------
// C[M,N] = act(A[M,K] @ Bt[N,K]^T + bias[N])
// Block tile 128x128, 256 threads = 8 waves (4 M x 2 N), wave tile 32x64.
// ACT: 0 none, 1 relu, 2 sigmoid.  OUTM: 0 bf16, 1 f32, 2 both.
template <int ACT, int OUTM>
__global__ __launch_bounds__(256) void gemm_wmma(
    const bf16_t* __restrict__ A, const bf16_t* __restrict__ Bt,
    const float* __restrict__ bias, bf16_t* __restrict__ outb,
    float* __restrict__ outf, int M, int N, int K) {
  __shared__ bf16_t smem[4 * 128 * LSTR];
  const int tid  = threadIdx.x;
  const int lane = tid & 31;
  const int l16  = lane & 15;
  const int half = lane >> 4;
  const int wave = tid >> 5;
  const int wn   = wave & 1;   // 0..1
  const int wm   = wave >> 1;  // 0..3
  const int bm = blockIdx.y * 128;
  const int bn = blockIdx.x * 128;

  v8f acc[2][4] = {};
  gemm_k_loop(A, Bt, K, bm, bn, smem, acc, tid);

#pragma unroll
  for (int ni = 0; ni < 4; ++ni) {
    const int nc = bn + 64 * wn + 16 * ni + l16;
    const float bs = bias[nc];
#pragma unroll
    for (int mi = 0; mi < 2; ++mi) {
#pragma unroll
      for (int r = 0; r < 8; ++r) {
        const int m = bm + 32 * wm + 16 * mi + r + half * 8;
        float v = acc[mi][ni][r] + bs;
        if constexpr (ACT == 1) v = v > 0.0f ? v : 0.0f;
        if constexpr (ACT == 2) v = 1.0f / (1.0f + __expf(-v));
        const size_t o = (size_t)m * N + nc;
        if constexpr (OUTM == 0 || OUTM == 2) outb[o] = (bf16_t)v;
        if constexpr (OUTM == 1 || OUTM == 2) outf[o] = v;
      }
    }
  }
}

// --------------------------- VQ argmin -------------------------------------
// For each row of Z, find argmax_n ( z.c_n - 0.5*||c_n||^2 )  (== argmin dist).
// One block handles 128 rows, looping over all 8192 codes in 128-wide tiles.
__global__ __launch_bounds__(256) void vq_argmin_kernel(
    const bf16_t* __restrict__ Zb, const bf16_t* __restrict__ Cb,
    const float* __restrict__ cnorm, int* __restrict__ out_idx) {
  __shared__ bf16_t smem[4 * 128 * LSTR];
  __shared__ float ls[256];
  __shared__ int   li[256];
  const int tid  = threadIdx.x;
  const int lane = tid & 31;
  const int l16  = lane & 15;
  const int half = lane >> 4;
  const int wave = tid >> 5;
  const int wn   = wave & 1;
  const int wm   = wave >> 1;
  const int bm   = blockIdx.x * 128;

  float best_s[2][8];
  int   best_n[2][8];
#pragma unroll
  for (int mi = 0; mi < 2; ++mi)
#pragma unroll
    for (int r = 0; r < 8; ++r) { best_s[mi][r] = -3.402823466e38f; best_n[mi][r] = 0; }

  for (int nt = 0; nt < NCODE; nt += 128) {
    v8f acc[2][4] = {};
    gemm_k_loop(Zb, Cb, LAT, bm, nt, smem, acc, tid);

#pragma unroll
    for (int ni = 0; ni < 4; ++ni) {
      const int n = nt + 64 * wn + 16 * ni + l16;
      const float hcn = 0.5f * cnorm[n];
#pragma unroll
      for (int mi = 0; mi < 2; ++mi) {
#pragma unroll
        for (int r = 0; r < 8; ++r) {
          float s = acc[mi][ni][r] - hcn;
          int nn = n;
#pragma unroll
          for (int off = 8; off >= 1; off >>= 1) {  // butterfly within 16-lane half
            float os = __shfl_xor(s, off, 32);
            int   on = __shfl_xor(nn, off, 32);
            if (os > s || (os == s && on < nn)) { s = os; nn = on; }
          }
          if (s > best_s[mi][r] || (s == best_s[mi][r] && nn < best_n[mi][r])) {
            best_s[mi][r] = s; best_n[mi][r] = nn;
          }
        }
      }
    }
  }
  // merge the two wave-columns via LDS
#pragma unroll
  for (int mi = 0; mi < 2; ++mi)
#pragma unroll
    for (int r = 0; r < 8; ++r) {
      const int rl = 32 * wm + 16 * mi + r + half * 8;  // 0..127
      if (l16 == 0) { ls[rl * 2 + wn] = best_s[mi][r]; li[rl * 2 + wn] = best_n[mi][r]; }
    }
  __syncthreads();
  if (tid < 128) {
    const float s0 = ls[tid * 2], s1 = ls[tid * 2 + 1];
    const int n0 = li[tid * 2], n1 = li[tid * 2 + 1];
    out_idx[bm + tid] = (s1 > s0 || (s1 == s0 && n1 < n0)) ? n1 : n0;
  }
}

// --------------------------- small helpers ---------------------------------
__global__ void cvt_bf16_kernel(const float* __restrict__ in,
                                bf16_t* __restrict__ out, size_t n) {
  size_t i = (size_t)blockIdx.x * blockDim.x + threadIdx.x;
  const size_t stride = (size_t)gridDim.x * blockDim.x;
  for (; i < n; i += stride) out[i] = (bf16_t)in[i];
}

// out[n*K + k] = bf16(in[k*N + n])  (in is [K,N])
__global__ void transpose_cvt_kernel(const float* __restrict__ in,
                                     bf16_t* __restrict__ out, int K, int N) {
  size_t i = (size_t)blockIdx.x * blockDim.x + threadIdx.x;
  const size_t total = (size_t)K * N;
  const size_t stride = (size_t)gridDim.x * blockDim.x;
  for (; i < total; i += stride) {
    const int k = (int)(i / N);
    const int n = (int)(i % N);
    out[(size_t)n * K + k] = (bf16_t)in[i];
  }
}

__global__ void cnorm_kernel(const float* __restrict__ cb, float* __restrict__ cn) {
  const int j = blockIdx.x * blockDim.x + threadIdx.x;
  if (j < NCODE) {
    float s = 0.0f;
    const float* p = cb + (size_t)j * LAT;
#pragma unroll 8
    for (int k = 0; k < LAT; ++k) { const float v = p[k]; s += v * v; }
    cn[j] = s;
  }
}

__global__ void zero1_kernel(float* p) { *p = 0.0f; }

// one block per row (256 threads == LAT): gather z_q (bf16 for decoder) and
// accumulate loss = 1.25 * mean((z_q - z)^2)
__global__ __launch_bounds__(256) void gather_loss_kernel(
    const int* __restrict__ idx, const float* __restrict__ cb,
    const float* __restrict__ zf, bf16_t* __restrict__ zqb,
    float* __restrict__ loss) {
  const int row = blockIdx.x;
  const int j = threadIdx.x;
  const int id = idx[row];
  const float c = cb[(size_t)id * LAT + j];
  const float zv = zf[(size_t)row * LAT + j];
  zqb[(size_t)row * LAT + j] = (bf16_t)c;
  const float d = c - zv;
  float v = d * d;
#pragma unroll
  for (int off = 16; off >= 1; off >>= 1) v += __shfl_xor(v, off, 32);
  __shared__ float red[8];
  const int lane = j & 31, w = j >> 5;
  if (lane == 0) red[w] = v;
  __syncthreads();
  if (j == 0) {
    float s = 0.0f;
#pragma unroll
    for (int i = 0; i < 8; ++i) s += red[i];
    atomicAdd(loss, s * (1.25f / ((float)BATCH * (float)LAT)));
  }
}

// --------------------------- launcher --------------------------------------
extern "C" void kernel_launch(void* const* d_in, const int* in_sizes, int n_in,
                              void* d_out, int out_size, void* d_ws, size_t ws_size,
                              hipStream_t stream) {
  const float* x        = (const float*)d_in[0];
  const float* enc_w1   = (const float*)d_in[1];  // [1024,2048]
  const float* enc_b1   = (const float*)d_in[2];
  const float* enc_w2   = (const float*)d_in[3];  // [2048,256]
  const float* enc_b2   = (const float*)d_in[4];
  const float* codebook = (const float*)d_in[5];  // [8192,256]
  const float* dec_w1   = (const float*)d_in[6];  // [256,2048]
  const float* dec_b1   = (const float*)d_in[7];
  const float* dec_w2   = (const float*)d_in[8];  // [2048,1024]
  const float* dec_b2   = (const float*)d_in[9];

  float* x_recon = (float*)d_out;
  float* d_loss  = x_recon + (size_t)BATCH * IN_DIM;

  // workspace carve-out (~134 MB total)
  char* p = (char*)d_ws;
  auto alloc = [&](size_t bytes) -> char* {
    char* r = p;
    p += (bytes + 255) & ~(size_t)255;
    return r;
  };
  bf16_t* xb    = (bf16_t*)alloc((size_t)BATCH * IN_DIM * sizeof(bf16_t)); // 32 MB
  bf16_t* Wt1   = (bf16_t*)alloc((size_t)HID * IN_DIM * sizeof(bf16_t));   //  4 MB
  bf16_t* Wt2   = (bf16_t*)alloc((size_t)LAT * HID * sizeof(bf16_t));      //  1 MB
  bf16_t* cbb   = (bf16_t*)alloc((size_t)NCODE * LAT * sizeof(bf16_t));    //  4 MB
  bf16_t* Wd1t  = (bf16_t*)alloc((size_t)HID * LAT * sizeof(bf16_t));      //  1 MB
  bf16_t* Wd2t  = (bf16_t*)alloc((size_t)IN_DIM * HID * sizeof(bf16_t));   //  4 MB
  bf16_t* hbuf  = (bf16_t*)alloc((size_t)BATCH * HID * sizeof(bf16_t));    // 64 MB (h, later hd)
  float*  zf    = (float*)alloc((size_t)BATCH * LAT * sizeof(float));      // 16 MB
  bf16_t* zb    = (bf16_t*)alloc((size_t)BATCH * LAT * sizeof(bf16_t));    //  8 MB (z bf16, later z_q)
  float*  cnorm = (float*)alloc((size_t)NCODE * sizeof(float));
  int*    idx   = (int*)alloc((size_t)BATCH * sizeof(int));

  // ---- prep: loss=0, bf16 conversions / weight transposes --------------------
  zero1_kernel<<<1, 1, 0, stream>>>(d_loss);
  cvt_bf16_kernel<<<4096, 256, 0, stream>>>(x, xb, (size_t)BATCH * IN_DIM);
  transpose_cvt_kernel<<<4096, 256, 0, stream>>>(enc_w1, Wt1, IN_DIM, HID);
  transpose_cvt_kernel<<<1024, 256, 0, stream>>>(enc_w2, Wt2, HID, LAT);
  cvt_bf16_kernel<<<2048, 256, 0, stream>>>(codebook, cbb, (size_t)NCODE * LAT);
  cnorm_kernel<<<NCODE / 256, 256, 0, stream>>>(codebook, cnorm);
  transpose_cvt_kernel<<<1024, 256, 0, stream>>>(dec_w1, Wd1t, LAT, HID);
  transpose_cvt_kernel<<<4096, 256, 0, stream>>>(dec_w2, Wd2t, HID, IN_DIM);

  // ---- encoder ---------------------------------------------------------------
  // h = relu(x @ W1 + b1) : [16384,2048] bf16
  gemm_wmma<1, 0><<<dim3(HID / 128, BATCH / 128), 256, 0, stream>>>(
      xb, Wt1, enc_b1, hbuf, nullptr, BATCH, HID, IN_DIM);
  // z = h @ W2 + b2 : f32 + bf16
  gemm_wmma<0, 2><<<dim3(LAT / 128, BATCH / 128), 256, 0, stream>>>(
      hbuf, Wt2, enc_b2, zb, zf, BATCH, LAT, HID);

  // ---- vector quantizer ------------------------------------------------------
  vq_argmin_kernel<<<BATCH / 128, 256, 0, stream>>>(zb, cbb, cnorm, idx);
  // z_q gather (bf16, reusing zb) + loss accumulation (f32)
  gather_loss_kernel<<<BATCH, 256, 0, stream>>>(idx, codebook, zf, zb, d_loss);

  // ---- decoder ---------------------------------------------------------------
  // hd = relu(z_q @ Wd1 + bd1) : reuse hbuf
  gemm_wmma<1, 0><<<dim3(HID / 128, BATCH / 128), 256, 0, stream>>>(
      zb, Wd1t, dec_b1, hbuf, nullptr, BATCH, HID, LAT);
  // x_recon = sigmoid(hd @ Wd2 + bd2) : f32 straight into d_out
  gemm_wmma<2, 1><<<dim3(IN_DIM / 128, BATCH / 128), 256, 0, stream>>>(
      hbuf, Wd2t, dec_b2, nullptr, x_recon, BATCH, IN_DIM, HID);
}
